// Attention_29755533427577
// MI455X (gfx1250) — matrix operand order repaired and checked by reading, not compiled, not run on hardware
//
#include <hip/hip_runtime.h>
#include <stdint.h>

#define S_ENC 2048
#define S_DEC 2048
#define DHEAD 512
#define BATCH 8
#define E_TILE 32
#define NW 8
#define WG_Q (NW * 16)          // 128 queries per workgroup
#define NTHREADS (NW * 32)      // 256 threads
#define N_DTILES (DHEAD / 16)   // 32
#define N_KSTEP  (DHEAD / 32)   // 16
#define N_TILES  (S_ENC / E_TILE) // 64

typedef __bf16 bf16;
typedef __attribute__((ext_vector_type(16))) __bf16 v16bf;
typedef __attribute__((ext_vector_type(8)))  float  v8f;
typedef __attribute__((ext_vector_type(4)))  float  f32x4;
typedef __attribute__((ext_vector_type(4)))  uint32_t u32x4;
typedef __attribute__((ext_vector_type(4)))  int      i32x4;

// LDS strides in bf16 elements (row byte-stride 16B-aligned, bank-spread)
#define LDA_STRIDE 520   // tileA: [E_TILE][.]  enc tile row-major [e][d]
#define LDB_STRIDE 40    // tileB: [DHEAD][.]   enc tile transposed [d][e]
#define LDP_STRIDE 40    // ldsP: per-wave [16][.] P tile row-major [m][k]

#define SZ_A  (E_TILE * LDA_STRIDE * 2)   // 33280
#define SZ_B  (DHEAD * LDB_STRIDE * 2)    // 40960
#define OFF_A0 0
#define OFF_B0 (OFF_A0 + SZ_A)
#define OFF_A1 (OFF_B0 + SZ_B)
#define OFF_B1 (OFF_A1 + SZ_A)
#define OFF_P  (OFF_B1 + SZ_B)
#define SZ_P   (NW * 16 * LDP_STRIDE * 2) // 10240
#define LDS_BYTES (OFF_P + SZ_P)          // 158720 -> fits 320KB WGP

#define ENC_ELEMS ((size_t)BATCH * S_ENC * DHEAD)

static __device__ __forceinline__ uint32_t pack2_bf16(float a, float b) {
    union { bf16 h[2]; uint32_t u; } x;
    x.h[0] = (bf16)a; x.h[1] = (bf16)b;
    return x.u;
}

static __device__ __forceinline__ u32x4 cvt8(const float* p) {
    f32x4 a = *(const f32x4*)(p);
    f32x4 b = *(const f32x4*)(p + 4);
    u32x4 r;
    r.x = pack2_bf16(a.x, a.y);
    r.y = pack2_bf16(a.z, a.w);
    r.z = pack2_bf16(b.x, b.y);
    r.w = pack2_bf16(b.z, b.w);
    return r;
}

static __device__ __forceinline__ v16bf lds_frag32(const bf16* p) {
    v16bf f;
    u32x4* v = (u32x4*)&f;
    v[0] = *(const u32x4*)(p);
    v[1] = *(const u32x4*)(p + 16);
    return f;
}

// ---- CDNA5 async global->LDS copy (ASYNCcnt), 16 bytes per lane ----
typedef __attribute__((address_space(1))) i32x4* g_v4i_ptr;
typedef __attribute__((address_space(3))) i32x4* l_v4i_ptr;

static __device__ __forceinline__ void async_cp16(const bf16* g, bf16* l) {
#if __has_builtin(__builtin_amdgcn_global_load_async_to_lds_b128)
    __builtin_amdgcn_global_load_async_to_lds_b128(
        (g_v4i_ptr)(uintptr_t)g,
        (l_v4i_ptr)(uint32_t)(uintptr_t)l,
        0, 0);
#else
    asm volatile("global_load_async_to_lds_b128 %0, %1, off"
                 :: "v"((uint32_t)(uintptr_t)l), "v"(g)
                 : "memory");
#endif
}

static __device__ __forceinline__ void wait_async0() {
#if __has_builtin(__builtin_amdgcn_s_wait_asynccnt)
    __builtin_amdgcn_s_wait_asynccnt(0);
#else
    asm volatile("s_wait_asynccnt 0x0" ::: "memory");
#endif
}

// =======================================================================
// Prep: enc f32 -> bf16, both row-major (encH) and transposed (encT)
// =======================================================================
__global__ void __launch_bounds__(256)
prep_enc_bf16(const float* __restrict__ enc,
              bf16* __restrict__ encH,   // [B][S_ENC][DHEAD]
              bf16* __restrict__ encT)   // [B][DHEAD][S_ENC]
{
    __shared__ bf16 tile[32][33];
    const int tid = threadIdx.x;
    const int d0 = blockIdx.x * 32;
    const int e0 = blockIdx.y * 32;
    const int b  = blockIdx.z;

    const float* src = enc + ((size_t)b * S_ENC + e0) * DHEAD + d0;
    bf16* dstH = encH + ((size_t)b * S_ENC + e0) * DHEAD + d0;
    bf16* dstT = encT + ((size_t)b * DHEAD + d0) * S_ENC + e0;

    {
        int e = tid >> 3, c = (tid & 7) * 4;
        f32x4 v = *(const f32x4*)(src + (size_t)e * DHEAD + c);
        tile[e][c + 0] = (bf16)v.x;
        tile[e][c + 1] = (bf16)v.y;
        tile[e][c + 2] = (bf16)v.z;
        tile[e][c + 3] = (bf16)v.w;
        uint2 p; p.x = pack2_bf16(v.x, v.y); p.y = pack2_bf16(v.z, v.w);
        *(uint2*)(dstH + (size_t)e * DHEAD + c) = p;
    }
    __syncthreads();
    {
        int d = tid >> 3, c = (tid & 7) * 4;
        union { bf16 h[4]; uint2 u; } o;
        o.h[0] = tile[c + 0][d];
        o.h[1] = tile[c + 1][d];
        o.h[2] = tile[c + 2][d];
        o.h[3] = tile[c + 3][d];
        *(uint2*)(dstT + (size_t)d * S_ENC + c) = o.u;
    }
}

// =======================================================================
// Main: flash attention, WMMA bf16, async double-buffered enc tiles
// =======================================================================
__global__ void __launch_bounds__(NTHREADS, 1)
attn_flash_wmma(const bf16* __restrict__ encH,
                const bf16* __restrict__ encT,
                const float* __restrict__ dec,
                float* __restrict__ out)
{
    extern __shared__ char lds_raw[];
    bf16* A0   = (bf16*)(lds_raw + OFF_A0);
    bf16* B0   = (bf16*)(lds_raw + OFF_B0);
    bf16* A1   = (bf16*)(lds_raw + OFF_A1);
    bf16* B1   = (bf16*)(lds_raw + OFF_B1);
    bf16* ldsP = (bf16*)(lds_raw + OFF_P);

    const int tid  = threadIdx.x;
    const int wave = tid >> 5;
    const int lane = tid & 31;
    const int lm   = lane & 15;
    const int hi   = lane >> 4;

    const int b  = blockIdx.y;
    const int tq = blockIdx.x * WG_Q + wave * 16;

    const bf16*  encHb = encH + (size_t)b * S_ENC * DHEAD;
    const bf16*  encTb = encT + (size_t)b * DHEAD * S_ENC;
    const float* decB  = dec  + (size_t)b * S_DEC * DHEAD;
    float*       outB  = out  + (size_t)b * S_DEC * DHEAD;

    // ---- Q A-fragments: global f32 -> bf16 registers, once ----
    v16bf Qf[N_KSTEP];
    {
        const float* qrow = decB + (size_t)(tq + lm) * DHEAD;
        #pragma unroll
        for (int k = 0; k < N_KSTEP; ++k) {
            u32x4* qv = (u32x4*)&Qf[k];
            qv[0] = cvt8(qrow + 32 * k + 8 * hi);
            qv[1] = cvt8(qrow + 32 * k + 16 + 8 * hi);
        }
    }

    v8f O[N_DTILES];
    #pragma unroll
    for (int n = 0; n < N_DTILES; ++n) O[n] = v8f{0.f,0.f,0.f,0.f,0.f,0.f,0.f,0.f};
    float m[8], l[8];
    #pragma unroll
    for (int r = 0; r < 8; ++r) { m[r] = -3.0e38f; l[r] = 0.f; }

    bf16* pw = ldsP + (size_t)wave * 16 * LDP_STRIDE;

    // issue async copy of one enc tile (this wave's 1/8 share: 16 instrs)
    auto issue_tile = [&](int e0, bf16* bufA, bf16* bufB) {
        const bf16* gA = encHb + (size_t)e0 * DHEAD;
        const bf16* gT = encTb + e0;
        #pragma unroll
        for (int t = 0; t < 8; ++t) {          // row-major tile: 32 rows x 1KB
            int i = (wave * 8 + t) * 32 + lane;
            int row = i >> 6, c = i & 63;      // 16B chunks
            async_cp16(gA + (size_t)row * DHEAD + c * 8,
                       bufA + row * LDA_STRIDE + c * 8);
        }
        #pragma unroll
        for (int t = 0; t < 8; ++t) {          // transposed tile: 512 rows x 64B
            int i = (wave * 8 + t) * 32 + lane;
            int d = i >> 2, c = i & 3;
            async_cp16(gT + (size_t)d * S_ENC + c * 8,
                       bufB + d * LDB_STRIDE + c * 8);
        }
    };

    // process one staged tile: GEMM1 + online softmax + P transpose + GEMM2
    auto process_tile = [&](const bf16* tileA, const bf16* tileB) {
        const bf16* aRow0 = tileA + lm * LDA_STRIDE + 8 * hi;
        const bf16* aRow1 = tileA + (16 + lm) * LDA_STRIDE + 8 * hi;
        const bf16* bRow  = tileB + lm * LDB_STRIDE + 8 * hi;

        v8f S0 = v8f{0.f,0.f,0.f,0.f,0.f,0.f,0.f,0.f};
        v8f S1 = v8f{0.f,0.f,0.f,0.f,0.f,0.f,0.f,0.f};
        v16bf b0 = lds_frag32(aRow0);
        v16bf b1 = lds_frag32(aRow1);
        #pragma unroll
        for (int k = 0; k < N_KSTEP; ++k) {
            v16bf nb0, nb1;
            if (k < N_KSTEP - 1) {
                nb0 = lds_frag32(aRow0 + 32 * (k + 1));
                nb1 = lds_frag32(aRow1 + 32 * (k + 1));
            }
            S0 = __builtin_amdgcn_wmma_f32_16x16x32_bf16(
                    false, Qf[k], false, b0, (short)0, S0, false, false);
            S1 = __builtin_amdgcn_wmma_f32_16x16x32_bf16(
                    false, Qf[k], false, b1, (short)0, S1, false, false);
            b0 = nb0; b1 = nb1;
        }

        float sc[8];
        #pragma unroll
        for (int r = 0; r < 8; ++r) {
            float v = fmaxf(S0[r], S1[r]);
            v = fmaxf(v, __shfl_xor(v, 1, 32));
            v = fmaxf(v, __shfl_xor(v, 2, 32));
            v = fmaxf(v, __shfl_xor(v, 4, 32));
            v = fmaxf(v, __shfl_xor(v, 8, 32));
            float mn = fmaxf(m[r], v);
            sc[r] = __expf(m[r] - mn);
            m[r] = mn;
        }
        #pragma unroll
        for (int r = 0; r < 8; ++r) {
            float p0 = __expf(S0[r] - m[r]);
            float p1 = __expf(S1[r] - m[r]);
            S0[r] = p0; S1[r] = p1;
            float s = p0 + p1;
            s += __shfl_xor(s, 1, 32);
            s += __shfl_xor(s, 2, 32);
            s += __shfl_xor(s, 4, 32);
            s += __shfl_xor(s, 8, 32);
            l[r] = l[r] * sc[r] + s;
        }
        #pragma unroll
        for (int n = 0; n < N_DTILES; ++n) {
            #pragma unroll
            for (int r = 0; r < 8; ++r) O[n][r] *= sc[r];
        }

        #pragma unroll
        for (int r = 0; r < 8; ++r) {
            pw[(r + 8 * hi) * LDP_STRIDE + lm]      = (bf16)S0[r];
            pw[(r + 8 * hi) * LDP_STRIDE + 16 + lm] = (bf16)S1[r];
        }
        asm volatile("" ::: "memory");   // LDS in-order per wave
        v16bf pa;
        {
            const bf16* pp = pw + lm * LDP_STRIDE + 8 * hi;
            u32x4* av = (u32x4*)&pa;
            av[0] = *(const u32x4*)(pp);
            av[1] = *(const u32x4*)(pp + 16);
        }

        v16bf bb = lds_frag32(bRow);
        #pragma unroll
        for (int n = 0; n < N_DTILES; ++n) {
            v16bf nb;
            if (n < N_DTILES - 1) nb = lds_frag32(bRow + (16 * (n + 1)) * LDB_STRIDE);
            O[n] = __builtin_amdgcn_wmma_f32_16x16x32_bf16(
                       false, pa, false, bb, (short)0, O[n], false, false);
            bb = nb;
        }
    };

    // ---- double-buffered async pipeline over encoder tiles ----
    issue_tile(0, A0, B0);
    #pragma unroll 1
    for (int it = 0; it < N_TILES; it += 2) {
        wait_async0();
        __syncthreads();
        issue_tile((it + 1) * E_TILE, A1, B1);   // DMA overlaps compute below
        process_tile(A0, B0);

        wait_async0();
        __syncthreads();
        if (it + 2 < N_TILES) issue_tile((it + 2) * E_TILE, A0, B0);
        process_tile(A1, B1);
    }

    // ---- epilogue: normalize and store ----
    float rl[8];
    #pragma unroll
    for (int r = 0; r < 8; ++r) rl[r] = 1.0f / l[r];

    #pragma unroll
    for (int n = 0; n < N_DTILES; ++n) {
        #pragma unroll
        for (int r = 0; r < 8; ++r) {
            int t = tq + r + 8 * hi;
            outB[(size_t)t * DHEAD + 16 * n + lm] = O[n][r] * rl[r];
        }
    }
}

extern "C" void kernel_launch(void* const* d_in, const int* in_sizes, int n_in,
                              void* d_out, int out_size, void* d_ws, size_t ws_size,
                              hipStream_t stream) {
    const float* enc = (const float*)d_in[0];
    const float* dec = (const float*)d_in[1];
    float* out = (float*)d_out;

    bf16* encH = (bf16*)d_ws;                          // 16 MB
    bf16* encT = (bf16*)((char*)d_ws + ENC_ELEMS * 2); // 16 MB

    dim3 pgrid(DHEAD / 32, S_ENC / 32, BATCH);
    prep_enc_bf16<<<pgrid, 256, 0, stream>>>(enc, encH, encT);

    (void)hipFuncSetAttribute((const void*)attn_flash_wmma,
                              hipFuncAttributeMaxDynamicSharedMemorySize, LDS_BYTES);
    dim3 grid(S_DEC / WG_Q, BATCH);
    attn_flash_wmma<<<grid, NTHREADS, LDS_BYTES, stream>>>(encH, encT, dec, out);
}
